// DeepSeekMoE_17892833755768
// MI455X (gfx1250) — matrix operand order repaired and checked by reading, not compiled, not run on hardware
//
#include <hip/hip_runtime.h>

// Problem constants: B=2, S=1024, D=2048, I=1408, E=8, TOPK=2
#define TKN   2048      // B*S tokens
#define DDIM  2048
#define IDIM  1408
#define NEXP  8
#define NE9   9         // shared expert + 8 routed
#define LDSK  40        // 32 K elements + 8 pad (f16) => 80B row stride
#define KSTEPS2 (NE9 * (IDIM / 32))   // 396 flattened K-iterations for GEMM2

typedef __attribute__((ext_vector_type(16))) _Float16 v16h;
typedef __attribute__((ext_vector_type(8)))  _Float16 v8h;
typedef __attribute__((ext_vector_type(4)))  _Float16 v4h;
typedef __attribute__((ext_vector_type(8)))  float    v8f;

// ---------------------------------------------------------------------------
// CDNA5 async global->LDS copy (GLOBAL_LOAD_ASYNC_TO_LDS_B128, ASYNCcnt).
// Builtin signature (from hipcc diagnostic): param0 = int32x4 in addrspace(1)
// (printed as "__device__"), so pass typed v4i pointers: (global src, lds dst).
// ---------------------------------------------------------------------------
#if __has_builtin(__builtin_amdgcn_global_load_async_to_lds_b128)
#define HAS_ASYNC_LDS 1
typedef int v4i __attribute__((vector_size(16)));
typedef __attribute__((address_space(1))) v4i* g_b128_p;
typedef __attribute__((address_space(3))) v4i* l_b128_p;
#else
#define HAS_ASYNC_LDS 0
#endif

__device__ __forceinline__ void copy16(_Float16* l, const _Float16* g) {
#if HAS_ASYNC_LDS
  __builtin_amdgcn_global_load_async_to_lds_b128(
      (g_b128_p)(v4i*)const_cast<_Float16*>(g),
      (l_b128_p)(v4i*)l, 0, 0);
#else
  *(v8h*)l = *(const v8h*)g;
#endif
}

__device__ __forceinline__ void wait_async_all() {
#if HAS_ASYNC_LDS
#if __has_builtin(__builtin_amdgcn_s_wait_asynccnt)
  __builtin_amdgcn_s_wait_asynccnt(0);
#else
  asm volatile("s_wait_asynccnt 0x0" ::: "memory");
#endif
#endif
}

// ---------------------------------------------------------------------------
// fp32 -> fp16 bulk convert (vectorized by 4)
// ---------------------------------------------------------------------------
__global__ __launch_bounds__(256)
void k_cvt_f32_f16(const float* __restrict__ src, _Float16* __restrict__ dst, int n4) {
  int i = blockIdx.x * blockDim.x + threadIdx.x;
  if (i >= n4) return;
  float4 v = ((const float4*)src)[i];
  v4h h;
  h[0] = (_Float16)v.x; h[1] = (_Float16)v.y;
  h[2] = (_Float16)v.z; h[3] = (_Float16)v.w;
  *(v4h*)(dst + (size_t)i * 4) = h;
}

// ---------------------------------------------------------------------------
// Router: sigmoid scores, top-2 (bias affects selection only), normalized
// weights from raw scores. Writes topk indices + dense combine table.
// ---------------------------------------------------------------------------
__global__ __launch_bounds__(256)
void k_router(const float* __restrict__ x, const float* __restrict__ gw,
              const float* __restrict__ bias, float* __restrict__ comb,
              int* __restrict__ idx_out) {
  int t = blockIdx.x * blockDim.x + threadIdx.x;   // one thread per token
  const float* xr = x + (size_t)t * DDIM;
  float acc[NEXP];
#pragma unroll
  for (int e = 0; e < NEXP; ++e) acc[e] = 0.0f;
  for (int d = 0; d < DDIM; ++d) {
    float xv = xr[d];
#pragma unroll
    for (int e = 0; e < NEXP; ++e) acc[e] += xv * gw[e * DDIM + d];
  }
  float s[NEXP];
#pragma unroll
  for (int e = 0; e < NEXP; ++e) s[e] = 1.0f / (1.0f + __expf(-acc[e]));
  int i0 = 0, i1 = -1;
  float b0 = s[0] + bias[0], b1 = -3.4e38f;
#pragma unroll
  for (int e = 1; e < NEXP; ++e) {
    float v = s[e] + bias[e];
    if (v > b0) { b1 = b0; i1 = i0; b0 = v; i0 = e; }
    else if (v > b1) { b1 = v; i1 = e; }
  }
  float w0 = s[i0], w1 = s[i1];
  float inv = 1.0f / (w0 + w1);
#pragma unroll
  for (int e = 0; e < NEXP; ++e) comb[t * NEXP + e] = 0.0f;
  comb[t * NEXP + i0] = w0 * inv;
  comb[t * NEXP + i1] = w1 * inv;
  idx_out[t * 2 + 0] = i0;
  idx_out[t * 2 + 1] = i1;
}

// ---------------------------------------------------------------------------
// WMMA fragment helpers (CDNA5 16x16x32 f16 layouts)
// ---------------------------------------------------------------------------
__device__ __forceinline__ v16h loadAfrag(const _Float16* rowBase, int khalf) {
  v8h lo = *(const v8h*)(rowBase + khalf * 16);
  v8h hi = *(const v8h*)(rowBase + khalf * 16 + 8);
  return __builtin_shufflevector(lo, hi, 0,1,2,3,4,5,6,7,8,9,10,11,12,13,14,15);
}
__device__ __forceinline__ v16h loadBfrag(const _Float16* p) {
  v8h lo = *(const v8h*)(p);
  v8h hi = *(const v8h*)(p + 8);
  return __builtin_shufflevector(lo, hi, 0,1,2,3,4,5,6,7,8,9,10,11,12,13,14,15);
}
__device__ __forceinline__ v8f wmma16(v16h a, v16h b, v8f c) {
  return __builtin_amdgcn_wmma_f32_16x16x32_f16(false, a, false, b, (short)0, c,
                                                false, false);
}

// ---------------------------------------------------------------------------
// GEMM1 + SwiGLU. Expert z (0 = shared, 1..8 = routed):
//   A'[z][t][i] = silu(X·Wg_zT) * (X·Wu_zT) * routeWeight(t,z)   (f16)
// Block tile M=128 x N=128, K-step 32, LDS double-buffered, async pipeline.
// 8 waves as 4(M) x 2(N); wave tile 32x64 => 16 WMMA per K-step per wave.
// ---------------------------------------------------------------------------
__global__ __launch_bounds__(256)
void k_gemm1_swiglu(const _Float16* __restrict__ Xb,
                    const _Float16* __restrict__ Wg,
                    const _Float16* __restrict__ Wu,
                    const float*    __restrict__ comb,
                    _Float16*       __restrict__ Aact) {
  __shared__ _Float16 lA[2][128 * LDSK];
  __shared__ _Float16 lG[2][128 * LDSK];
  __shared__ _Float16 lU[2][128 * LDSK];

  const int z     = blockIdx.z;
  const int mBase = blockIdx.y * 128;
  const int nBase = blockIdx.x * 128;

  const int tid  = threadIdx.x;
  const int lane = tid & 31;
  const int wave = tid >> 5;
  const int wm   = wave >> 1;   // 0..3
  const int wn   = wave & 1;    // 0..1

  const _Float16* wgE = Wg + (size_t)z * IDIM * DDIM;
  const _Float16* wuE = Wu + (size_t)z * IDIM * DDIM;

  v8f accG[2][4] = {};
  v8f accU[2][4] = {};

  const int am  = lane & 15;
  const int akh = lane >> 4;
  const int bn  = lane & 15;
  const int bkb = (lane >> 4) * 16;

  // Tile loader: 128x32 tile = 512 x 16B segments; 2 per thread.
  const int r0 = tid >> 2;           // rows 0..63 (and +64)
  const int c0 = (tid & 3) * 8;      // 0,8,16,24

  auto issue = [&](int k0, int b) {
    copy16(&lA[b][r0 * LDSK + c0],
           &Xb[(size_t)(mBase + r0) * DDIM + k0 + c0]);
    copy16(&lA[b][(r0 + 64) * LDSK + c0],
           &Xb[(size_t)(mBase + r0 + 64) * DDIM + k0 + c0]);
    copy16(&lG[b][r0 * LDSK + c0],
           &wgE[(size_t)(nBase + r0) * DDIM + k0 + c0]);
    copy16(&lG[b][(r0 + 64) * LDSK + c0],
           &wgE[(size_t)(nBase + r0 + 64) * DDIM + k0 + c0]);
    copy16(&lU[b][r0 * LDSK + c0],
           &wuE[(size_t)(nBase + r0) * DDIM + k0 + c0]);
    copy16(&lU[b][(r0 + 64) * LDSK + c0],
           &wuE[(size_t)(nBase + r0 + 64) * DDIM + k0 + c0]);
  };

  int buf = 0;
  issue(0, 0);
  for (int k0 = 0; k0 < DDIM; k0 += 32) {
    wait_async_all();
    __syncthreads();
    if (k0 + 32 < DDIM) issue(k0 + 32, buf ^ 1);   // overlaps WMMA below

    v16h aF0 = loadAfrag(&lA[buf][(wm * 32 + am) * LDSK], akh);
    v16h aF1 = loadAfrag(&lA[buf][(wm * 32 + 16 + am) * LDSK], akh);
#pragma unroll
    for (int j = 0; j < 4; ++j) {
      const int nrow = wn * 64 + j * 16 + bn;
      v16h gF = loadBfrag(&lG[buf][nrow * LDSK + bkb]);
      accG[0][j] = wmma16(aF0, gF, accG[0][j]);
      accG[1][j] = wmma16(aF1, gF, accG[1][j]);
      v16h uF = loadBfrag(&lU[buf][nrow * LDSK + bkb]);
      accU[0][j] = wmma16(aF0, uF, accU[0][j]);
      accU[1][j] = wmma16(aF1, uF, accU[1][j]);
    }
    __syncthreads();
    buf ^= 1;
  }

  // Epilogue: silu(g)*u * routeWeight -> f16 activations.
  const int rowHalf = (lane >> 4) * 8;
#pragma unroll
  for (int mi = 0; mi < 2; ++mi) {
#pragma unroll
    for (int r = 0; r < 8; ++r) {
      const int tRow = mBase + wm * 32 + mi * 16 + rowHalf + r;
      const float scale = (z == 0) ? 1.0f : comb[tRow * NEXP + (z - 1)];
#pragma unroll
      for (int j = 0; j < 4; ++j) {
        const int col = nBase + wn * 64 + j * 16 + (lane & 15);
        float g = accG[mi][j][r];
        float u = accU[mi][j][r];
        float sg = g / (1.0f + __expf(-g));
        Aact[((size_t)z * TKN + tRow) * IDIM + col] = (_Float16)(sg * u * scale);
      }
    }
  }
}

// ---------------------------------------------------------------------------
// GEMM2: out[t][d] = sum_{e=0..8} A'_e[t][:] . Wd_e[d][:]  (K = 9*I = 12672)
// Flattened (expert, k) pipeline so double-buffering runs uninterrupted.
// ---------------------------------------------------------------------------
__global__ __launch_bounds__(256)
void k_gemm2_combine(const _Float16* __restrict__ Aact,
                     const _Float16* __restrict__ Wd,
                     float*          __restrict__ out) {
  __shared__ _Float16 lA[2][128 * LDSK];
  __shared__ _Float16 lB[2][128 * LDSK];

  const int mBase = blockIdx.y * 128;   // tokens
  const int nBase = blockIdx.x * 128;   // D columns

  const int tid  = threadIdx.x;
  const int lane = tid & 31;
  const int wave = tid >> 5;
  const int wm   = wave >> 1;
  const int wn   = wave & 1;

  v8f acc[2][4] = {};

  const int am  = lane & 15;
  const int akh = lane >> 4;
  const int bn  = lane & 15;
  const int bkb = (lane >> 4) * 16;

  const int r0 = tid >> 2;
  const int c0 = (tid & 3) * 8;

  auto issue = [&](int it, int b) {
    const int e  = it / (IDIM / 32);
    const int k0 = (it % (IDIM / 32)) * 32;
    const _Float16* aE = Aact + (size_t)e * TKN * IDIM;
    const _Float16* wE = Wd   + (size_t)e * DDIM * IDIM;
    copy16(&lA[b][r0 * LDSK + c0],
           &aE[(size_t)(mBase + r0) * IDIM + k0 + c0]);
    copy16(&lA[b][(r0 + 64) * LDSK + c0],
           &aE[(size_t)(mBase + r0 + 64) * IDIM + k0 + c0]);
    copy16(&lB[b][r0 * LDSK + c0],
           &wE[(size_t)(nBase + r0) * IDIM + k0 + c0]);
    copy16(&lB[b][(r0 + 64) * LDSK + c0],
           &wE[(size_t)(nBase + r0 + 64) * IDIM + k0 + c0]);
  };

  int buf = 0;
  issue(0, 0);
  for (int it = 0; it < KSTEPS2; ++it) {
    wait_async_all();
    __syncthreads();
    if (it + 1 < KSTEPS2) issue(it + 1, buf ^ 1);

    v16h aF0 = loadAfrag(&lA[buf][(wm * 32 + am) * LDSK], akh);
    v16h aF1 = loadAfrag(&lA[buf][(wm * 32 + 16 + am) * LDSK], akh);
#pragma unroll
    for (int j = 0; j < 4; ++j) {
      const int nrow = wn * 64 + j * 16 + bn;
      v16h bF = loadBfrag(&lB[buf][nrow * LDSK + bkb]);
      acc[0][j] = wmma16(aF0, bF, acc[0][j]);
      acc[1][j] = wmma16(aF1, bF, acc[1][j]);
    }
    __syncthreads();
    buf ^= 1;
  }

  const int rowHalf = (lane >> 4) * 8;
#pragma unroll
  for (int mi = 0; mi < 2; ++mi) {
#pragma unroll
    for (int r = 0; r < 8; ++r) {
      const int tRow = mBase + wm * 32 + mi * 16 + rowHalf + r;
#pragma unroll
      for (int j = 0; j < 4; ++j) {
        const int col = nBase + wn * 64 + j * 16 + (lane & 15);
        out[(size_t)tRow * DDIM + col] = acc[mi][j][r];
      }
    }
  }
}

// ---------------------------------------------------------------------------
// Host-side orchestration
// ---------------------------------------------------------------------------
extern "C" void kernel_launch(void* const* d_in, const int* in_sizes, int n_in,
                              void* d_out, int out_size, void* d_ws, size_t ws_size,
                              hipStream_t stream) {
  (void)in_sizes; (void)n_in; (void)out_size; (void)ws_size;
  const float* x           = (const float*)d_in[0];
  const float* gate_w      = (const float*)d_in[1];
  const float* expert_bias = (const float*)d_in[2];
  const float* shared_gate = (const float*)d_in[3];
  const float* shared_up   = (const float*)d_in[4];
  const float* shared_down = (const float*)d_in[5];
  const float* exp_gate    = (const float*)d_in[6];
  const float* exp_up      = (const float*)d_in[7];
  const float* exp_down    = (const float*)d_in[8];

  const size_t nXb = (size_t)TKN * DDIM;
  const size_t nW  = (size_t)NE9 * IDIM * DDIM;
  const size_t nA  = (size_t)NE9 * TKN * IDIM;

  char* ws = (char*)d_ws;
  _Float16* Xb   = (_Float16*)ws;  ws += nXb * sizeof(_Float16);
  _Float16* Wg   = (_Float16*)ws;  ws += nW  * sizeof(_Float16);
  _Float16* Wu   = (_Float16*)ws;  ws += nW  * sizeof(_Float16);
  _Float16* Wdn  = (_Float16*)ws;  ws += nW  * sizeof(_Float16);
  _Float16* Aact = (_Float16*)ws;  ws += nA  * sizeof(_Float16);
  float*    comb = (float*)ws;

  float* out     = (float*)d_out;
  int*   idx_out = (int*)d_out + (size_t)TKN * DDIM;

  const size_t nSW = (size_t)IDIM * DDIM;
  const size_t nEW = (size_t)NEXP * IDIM * DDIM;

  auto cvt = [&](const float* s, _Float16* d, size_t n) {
    int n4 = (int)(n / 4);
    k_cvt_f32_f16<<<(n4 + 255) / 256, 256, 0, stream>>>(s, d, n4);
  };
  cvt(x, Xb, nXb);
  cvt(shared_gate, Wg,        nSW);
  cvt(exp_gate,    Wg + nSW,  nEW);
  cvt(shared_up,   Wu,        nSW);
  cvt(exp_up,      Wu + nSW,  nEW);
  cvt(shared_down, Wdn,       nSW);
  cvt(exp_down,    Wdn + nSW, nEW);

  k_router<<<TKN / 256, 256, 0, stream>>>(x, gate_w, expert_bias, comb, idx_out);

  k_gemm1_swiglu<<<dim3(IDIM / 128, TKN / 128, NE9), 256, 0, stream>>>(
      Xb, Wg, Wu, comb, Aact);

  k_gemm2_combine<<<dim3(DDIM / 128, TKN / 128), 256, 0, stream>>>(
      Aact, Wdn, out);
}
